// MiniKDALayer_1262720385766
// MI455X (gfx1250) — matrix operand order repaired and checked by reading, not compiled, not run on hardware
//
#include <hip/hip_runtime.h>
#include <math.h>

// ---------------------------------------------------------------------------
// MiniKDA layer for MI455X (gfx1250, wave32, WMMA).
// GEMMs: v_wmma_f32_16x16x32_f16, f16 inputs, f32 accumulate, fused epilogues,
//        software-pipelined K loop (next-slice fragments preloaded).
// Scan: single resident workgroup, state in VGPRs, LDS staging + reductions.
// ---------------------------------------------------------------------------

typedef __attribute__((ext_vector_type(16))) _Float16 v16h;
typedef __attribute__((ext_vector_type(8)))  _Float16 v8h;
typedef __attribute__((ext_vector_type(8)))  float    v8f;

#define RMS_EPS 1.1920929e-07f

__device__ __forceinline__ float sigmoidf_(float x) { return 1.0f / (1.0f + __expf(-x)); }
__device__ __forceinline__ float siluf_(float x)    { return x * sigmoidf_(x); }

// -------------------------------- converts --------------------------------
__global__ void cvt_f32_f16(const float* __restrict__ in, _Float16* __restrict__ out, int n) {
  int i = blockIdx.x * blockDim.x + threadIdx.x;
  if (i < n) out[i] = (_Float16)in[i];
}

// ------------------------------- WMMA GEMM --------------------------------
// C(MxN) = A(MxK) * B(NxK)^T.  A,B f16 row-major.  Epilogue via template EP.
enum { EP_F32 = 0, EP_SILU_F16 = 1, EP_MUL_F16 = 2, EP_RES_F32 = 3,
       EP_SIG_F32 = 4, EP_SILU_F32 = 5 };

template <int EP>
__device__ __forceinline__ void store_tile(v8f acc, int m0, int hl, int n, int N,
                                           float* __restrict__ Cf,
                                           _Float16* __restrict__ Ch,
                                           const float* __restrict__ res,
                                           const _Float16* __restrict__ mul) {
#pragma unroll
  for (int j = 0; j < 8; ++j) {
    int m = m0 + 8 * hl + j;
    size_t idx = (size_t)m * N + n;
    float x = acc[j];
    if constexpr (EP == EP_F32)       Cf[idx] = x;
    else if constexpr (EP == EP_SILU_F16) Ch[idx] = (_Float16)siluf_(x);
    else if constexpr (EP == EP_MUL_F16)  Ch[idx] = (_Float16)(x * (float)mul[idx]);
    else if constexpr (EP == EP_RES_F32)  Cf[idx] = x + res[idx];
    else if constexpr (EP == EP_SIG_F32)  Cf[idx] = sigmoidf_(x);
    else if constexpr (EP == EP_SILU_F32) Cf[idx] = siluf_(x);
  }
}

template <int EP>
__global__ __launch_bounds__(256) void gemm_wmma(
    const _Float16* __restrict__ A, const _Float16* __restrict__ B,
    int M, int N, int K,
    float* __restrict__ Cf, _Float16* __restrict__ Ch,
    const float* __restrict__ res, const _Float16* __restrict__ mul) {
  const int lane = threadIdx.x & 31;
  const int wave = threadIdx.x >> 5;
  const int hl   = lane >> 4;   // half-wave select
  const int l16  = lane & 15;
  const int m0 = (blockIdx.x * 8 + wave) * 16;   // 16-row strip per wave
  const int n0 = blockIdx.y * 64;                // 64 output cols per wave

  const _Float16* arow  = A + (size_t)(m0 + l16) * K;
  const _Float16* bbase = B + (size_t)(n0 + l16) * K + 16 * hl;

  // A fragment (16x32 f16): lane M = lane&15; K = 16h + [0..15] contiguous
  auto loadA = [&](int kk) -> v16h {
    return *(const v16h*)(arow + kk + 16 * hl);
  };
  // B fragment (32x16 f16): lane N = lane&15; K = 16h + [0..15] contiguous
  auto loadB = [&](int kk, int s) -> v16h {
    return *(const v16h*)(bbase + kk + (size_t)(16 * s) * K);
  };

  v8f acc0 = {}, acc1 = {}, acc2 = {}, acc3 = {};
  v16h a  = loadA(0);
  v16h b0 = loadB(0, 0), b1 = loadB(0, 1), b2 = loadB(0, 2), b3 = loadB(0, 3);

  for (int kk = 0; kk < K; kk += 32) {
    int kn = kk + 32;
    if (kn >= K) kn = 0;                          // wrap keeps tail load in-bounds
    __builtin_prefetch(arow + kn, 0, 3);          // global_prefetch, unconditional

    // issue next-slice loads before consuming current-slice fragments
    v16h an  = loadA(kn);
    v16h b0n = loadB(kn, 0);
    v16h b1n = loadB(kn, 1);
    v16h b2n = loadB(kn, 2);
    v16h b3n = loadB(kn, 3);

    acc0 = __builtin_amdgcn_wmma_f32_16x16x32_f16(false, a, false, b0, (short)0, acc0, false, false);
    acc1 = __builtin_amdgcn_wmma_f32_16x16x32_f16(false, a, false, b1, (short)0, acc1, false, false);
    acc2 = __builtin_amdgcn_wmma_f32_16x16x32_f16(false, a, false, b2, (short)0, acc2, false, false);
    acc3 = __builtin_amdgcn_wmma_f32_16x16x32_f16(false, a, false, b3, (short)0, acc3, false, false);

    a = an; b0 = b0n; b1 = b1n; b2 = b2n; b3 = b3n;
  }

  store_tile<EP>(acc0, m0, hl, n0 + 0  + l16, N, Cf, Ch, res, mul);
  store_tile<EP>(acc1, m0, hl, n0 + 16 + l16, N, Cf, Ch, res, mul);
  store_tile<EP>(acc2, m0, hl, n0 + 32 + l16, N, Cf, Ch, res, mul);
  store_tile<EP>(acc3, m0, hl, n0 + 48 + l16, N, Cf, Ch, res, mul);
}

// ------------------------------ beta kernel -------------------------------
__global__ void beta_kernel(const float* __restrict__ x, const float* __restrict__ Wb,
                            float* __restrict__ beta, int T, int d) {
  int lane = threadIdx.x & 31, wave = threadIdx.x >> 5;
  int t = blockIdx.x * 8 + wave;
  if (t >= T) return;
  const float* row = x + (size_t)t * d;
  float s = 0.f;
  for (int j = lane; j < d; j += 32) s += row[j] * Wb[j];
  for (int off = 16; off; off >>= 1) s += __shfl_down(s, off, 32);
  if (lane == 0) beta[t] = sigmoidf_(s);
}

// ------------------------------ PoPE kernel -------------------------------
__global__ void pope_kernel(const float* __restrict__ qpre, const float* __restrict__ kpre,
                            const float* __restrict__ delta,
                            float* __restrict__ q, float* __restrict__ k, int T, int dk) {
  int i = blockIdx.x * blockDim.x + threadIdx.x;
  if (i >= T * dk) return;
  int t = i / dk, j = i - t * dk;
  float freq = __expf((float)j * (logf(10000.0f) / (float)dk));
  float phi = (float)t * freq;
  size_t rowo = (size_t)t * 2 * dk;

  float xq = qpre[i];
  float muq = (xq > 20.f) ? xq : log1pf(__expf(xq));
  q[rowo + j]      = muq * cosf(phi);
  q[rowo + dk + j] = muq * sinf(phi);

  float phik = phi - 6.28318530717958647692f * sigmoidf_(delta[j]);
  float xk = kpre[i];
  float muk = (xk > 20.f) ? xk : log1pf(__expf(xk));
  k[rowo + j]      = muk * cosf(phik);
  k[rowo + dk + j] = muk * sinf(phik);
}

// ------------------------------- scan kernel ------------------------------
// Single workgroup (256 threads = 8 waves). State S[128][64] lives in VGPRs:
// thread (rg, c) with rg = tid/64 owns rows rg*32..rg*32+31 of column c.
__global__ __launch_bounds__(256) void scan_kernel(
    const float* __restrict__ q, const float* __restrict__ k,
    const float* __restrict__ v, const float* __restrict__ alpha,
    const float* __restrict__ beta, float* __restrict__ o, int T) {
  __shared__ float sh_a[128], sh_k[128], sh_q[128], sh_v[64], sh_b[1];
  __shared__ float red[4][64], ored[4][64];
  const int tid = threadIdx.x;
  const int c = tid & 63, rg = tid >> 6, r0 = rg * 32;
  float S[32];
#pragma unroll
  for (int i = 0; i < 32; ++i) S[i] = 0.f;

  for (int t = 0; t < T; ++t) {
    const size_t b128 = (size_t)t * 128;
    if (tid < 128) {
      sh_a[tid] = alpha[b128 + tid];
      sh_k[tid] = k[b128 + tid];
      sh_q[tid] = q[b128 + tid];
      // pull next step's vectors toward the WGP while this step computes
      __builtin_prefetch(alpha + b128 + 128 + tid, 0, 3);
      __builtin_prefetch(k + b128 + 128 + tid, 0, 3);
      __builtin_prefetch(q + b128 + 128 + tid, 0, 3);
    } else if (tid < 192) {
      sh_v[tid - 128] = v[(size_t)t * 64 + (tid - 128)];
      __builtin_prefetch(v + (size_t)t * 64 + (tid - 128) + 64, 0, 3);
    } else if (tid == 192) {
      sh_b[0] = beta[t];
    }
    __syncthreads();

    float part = 0.f;
#pragma unroll
    for (int i = 0; i < 32; ++i) part += sh_k[r0 + i] * sh_a[r0 + i] * S[i];
    red[rg][c] = part;
    __syncthreads();

    float ktaS = red[0][c] + red[1][c] + red[2][c] + red[3][c];
    float bt = sh_b[0], vt = sh_v[c];
    float op = 0.f;
#pragma unroll
    for (int i = 0; i < 32; ++i) {
      float s = sh_a[r0 + i] * S[i] + bt * sh_k[r0 + i] * (vt - ktaS);
      S[i] = s;
      op += s * sh_q[r0 + i];
    }
    ored[rg][c] = op;
    __syncthreads();

    if (rg == 0) o[(size_t)t * 64 + c] = ored[0][c] + ored[1][c] + ored[2][c] + ored[3][c];
    // next iteration's staging writes only begin after all threads pass the
    // barrier above; red reads done before it, ored reads before next step's first one -> 3 barriers/step.
  }
}

// ----------------------- RMSNorm(dv=64) * gate -> f16 ---------------------
__global__ void rmsnorm_gate_kernel(const float* __restrict__ o, const float* __restrict__ w,
                                    const float* __restrict__ gsig,
                                    _Float16* __restrict__ out, int T) {
  int lane = threadIdx.x & 31, wave = threadIdx.x >> 5;
  int t = blockIdx.x * 8 + wave;
  if (t >= T) return;
  const float* row = o + (size_t)t * 64;
  float v0 = row[lane], v1 = row[lane + 32];
  float ss = v0 * v0 + v1 * v1;
  for (int off = 16; off; off >>= 1) ss += __shfl_xor(ss, off, 32);
  float scale = rsqrtf(ss * (1.0f / 64.0f) + RMS_EPS);
  size_t b = (size_t)t * 64;
  out[b + lane]      = (_Float16)(v0 * scale * w[lane]      * gsig[b + lane]);
  out[b + lane + 32] = (_Float16)(v1 * scale * w[lane + 32] * gsig[b + lane + 32]);
}

// ----------------------- RMSNorm(d=1024) -> f16 row -----------------------
__global__ __launch_bounds__(256) void rmsnorm_row_kernel(
    const float* __restrict__ x, const float* __restrict__ w,
    _Float16* __restrict__ out, int d) {
  int t = blockIdx.x;
  const float* row = x + (size_t)t * d;
  __shared__ float sred[8];
  float ss = 0.f;
  for (int j = threadIdx.x; j < d; j += 256) { float v = row[j]; ss += v * v; }
  for (int off = 16; off; off >>= 1) ss += __shfl_xor(ss, off, 32);
  int lane = threadIdx.x & 31, wave = threadIdx.x >> 5;
  if (lane == 0) sred[wave] = ss;
  __syncthreads();
  if (threadIdx.x == 0) {
    float s = 0.f;
#pragma unroll
    for (int i = 0; i < 8; ++i) s += sred[i];
    sred[0] = rsqrtf(s / (float)d + RMS_EPS);
  }
  __syncthreads();
  float scale = sred[0];
  for (int j = threadIdx.x; j < d; j += 256)
    out[(size_t)t * d + j] = (_Float16)(row[j] * scale * w[j]);
}

// -------------------------------- launcher --------------------------------
extern "C" void kernel_launch(void* const* d_in, const int* in_sizes, int n_in,
                              void* d_out, int out_size, void* d_ws, size_t ws_size,
                              hipStream_t stream) {
  const float* x_seq  = (const float*)d_in[0];
  const float* W_q    = (const float*)d_in[1];
  const float* W_k    = (const float*)d_in[2];
  const float* W_v    = (const float*)d_in[3];
  const float* pdelta = (const float*)d_in[4];
  const float* W_a1   = (const float*)d_in[5];
  const float* W_a2   = (const float*)d_in[6];
  const float* W_beta = (const float*)d_in[7];
  const float* pnw    = (const float*)d_in[8];
  const float* W_og   = (const float*)d_in[9];
  const float* W_out  = (const float*)d_in[10];
  const float* fnw    = (const float*)d_in[11];
  const float* W_g    = (const float*)d_in[12];
  const float* W_u    = (const float*)d_in[13];
  const float* W_dn   = (const float*)d_in[14];
  float* out = (float*)d_out;

  const int d   = in_sizes[7];          // 1024
  const int dk  = in_sizes[4];          // 64
  const int dv  = in_sizes[8];          // 64
  const int dkp = 2 * dk;               // 128
  const int T   = in_sizes[0] / d;      // 8192

  // workspace carve-out (256B aligned)
  char* ws = (char*)d_ws;
  size_t off = 0;
  auto alloc = [&](size_t bytes) -> char* {
    char* p = ws + off;
    off = (off + bytes + 255) & ~(size_t)255;
    return p;
  };
  _Float16* X16   = (_Float16*)alloc((size_t)T * d * 2);      // x_seq f16, later h f16
  _Float16* Wq16  = (_Float16*)alloc((size_t)dk * d * 2);
  _Float16* Wk16  = (_Float16*)alloc((size_t)dk * d * 2);
  _Float16* Wv16  = (_Float16*)alloc((size_t)dv * d * 2);
  _Float16* Wa116 = (_Float16*)alloc((size_t)dkp * d * 2);
  _Float16* Wa216 = (_Float16*)alloc((size_t)dkp * dkp * 2);
  _Float16* Wog16 = (_Float16*)alloc((size_t)dv * d * 2);
  _Float16* Wo16  = (_Float16*)alloc((size_t)d * dv * 2);
  _Float16* Wg16  = (_Float16*)alloc((size_t)d * d * 2);
  _Float16* Wu16  = (_Float16*)alloc((size_t)d * d * 2);
  _Float16* Wd16  = (_Float16*)alloc((size_t)d * d * 2);
  float*    qpre  = (float*)alloc((size_t)T * dk * 4);
  float*    kpre  = (float*)alloc((size_t)T * dk * 4);
  float*    vbuf  = (float*)alloc((size_t)T * dv * 4);
  _Float16* a116  = (_Float16*)alloc((size_t)T * dkp * 2);
  float*    alpha = (float*)alloc((size_t)T * dkp * 4);
  float*    betab = (float*)alloc((size_t)T * 4);
  float*    qro   = (float*)alloc((size_t)T * dkp * 4);
  float*    kro   = (float*)alloc((size_t)T * dkp * 4);
  float*    obuf  = (float*)alloc((size_t)T * dv * 4);
  float*    gsig  = (float*)alloc((size_t)T * dv * 4);
  _Float16* att16 = (_Float16*)alloc((size_t)T * dv * 2);
  _Float16* G16   = (_Float16*)alloc((size_t)T * d * 2);
  _Float16* U16   = (_Float16*)alloc((size_t)T * d * 2);
  (void)ws_size; (void)n_in; (void)out_size;

  auto cvt = [&](const float* src, _Float16* dst, int n) {
    cvt_f32_f16<<<dim3((n + 255) / 256), dim3(256), 0, stream>>>(src, dst, n);
  };
  cvt(x_seq, X16, T * d);
  cvt(W_q,  Wq16,  dk * d);
  cvt(W_k,  Wk16,  dk * d);
  cvt(W_v,  Wv16,  dv * d);
  cvt(W_a1, Wa116, dkp * d);
  cvt(W_a2, Wa216, dkp * dkp);
  cvt(W_og, Wog16, dv * d);
  cvt(W_out, Wo16, d * dv);
  cvt(W_g,  Wg16,  d * d);
  cvt(W_u,  Wu16,  d * d);
  cvt(W_dn, Wd16,  d * d);

#define GEMM(EP, A, B, M, N, K, CF, CH, RES, MUL)                                \
  gemm_wmma<EP><<<dim3((M) / 128, (N) / 64), dim3(256), 0, stream>>>(            \
      A, B, M, N, K, CF, CH, RES, MUL)

  // projections
  GEMM(EP_F32,      X16, Wq16,  T, dk,  d,   qpre,  (_Float16*)nullptr, (const float*)nullptr, (const _Float16*)nullptr);
  GEMM(EP_F32,      X16, Wk16,  T, dk,  d,   kpre,  (_Float16*)nullptr, (const float*)nullptr, (const _Float16*)nullptr);
  GEMM(EP_SILU_F32, X16, Wv16,  T, dv,  d,   vbuf,  (_Float16*)nullptr, (const float*)nullptr, (const _Float16*)nullptr);
  GEMM(EP_SILU_F16, X16, Wa116, T, dkp, d,   (float*)nullptr, a116, (const float*)nullptr, (const _Float16*)nullptr);
  GEMM(EP_SIG_F32,  a116, Wa216, T, dkp, dkp, alpha, (_Float16*)nullptr, (const float*)nullptr, (const _Float16*)nullptr);
  beta_kernel<<<dim3(T / 8), dim3(256), 0, stream>>>(x_seq, W_beta, betab, T, d);
  pope_kernel<<<dim3((T * dk + 255) / 256), dim3(256), 0, stream>>>(qpre, kpre, pdelta, qro, kro, T, dk);

  // sequential gated delta-rule scan (latency wall; single resident WG)
  scan_kernel<<<dim3(1), dim3(256), 0, stream>>>(qro, kro, vbuf, alpha, betab, obuf, T);

  // output gate + post-norm + output projection (+ residual with x_seq)
  GEMM(EP_SIG_F32, X16, Wog16, T, dv, d, gsig, (_Float16*)nullptr, (const float*)nullptr, (const _Float16*)nullptr);
  rmsnorm_gate_kernel<<<dim3(T / 8), dim3(256), 0, stream>>>(obuf, pnw, gsig, att16, T);
  GEMM(EP_RES_F32, att16, Wo16, T, d, dv, out, (_Float16*)nullptr, x_seq, (const _Float16*)nullptr);

  // FFN: h = rmsnorm(out); out += (silu(h Wg^T) * (h Wu^T)) Wd^T
  rmsnorm_row_kernel<<<dim3(T), dim3(256), 0, stream>>>(out, fnw, X16, d);  // reuse X16 as h16
  GEMM(EP_SILU_F16, X16, Wg16, T, d, d, (float*)nullptr, G16, (const float*)nullptr, (const _Float16*)nullptr);
  GEMM(EP_MUL_F16,  X16, Wu16, T, d, d, (float*)nullptr, U16, (const float*)nullptr, G16);
  GEMM(EP_RES_F32,  U16, Wd16, T, d, d, out, (_Float16*)nullptr, out, (const _Float16*)nullptr);
#undef GEMM
}
